// ApproxMultLayer_11785390260575
// MI455X (gfx1250) — 8-bit matrix operand order repaired and checked by reading, compiled, not run on hardware
//
#include <hip/hip_runtime.h>
#include <hip/hip_bf16.h>
#include <stdint.h>

__device__ __forceinline__ void badd4(int a, int b, int c, int d, int cin,
                                      int& co, int& s) {
  int t = a + b + c + d + cin;
  co = t >> 1;
  s = t & 1;
}
__device__ __forceinline__ void badd2(int a, int b, int cin, int& co, int& s) {
  int t = a + b + cin;
  co = t >> 1;
  s = t & 1;
}
__device__ __forceinline__ void comp1(int a, int b, int c, int d, int& s, int& co) {
  int p1 = a ^ b, p2 = c ^ d;
  s = p1 ^ p2;
  co = (p1 & p2) | (a & b);
}
__device__ __forceinline__ int comp2(int a, int b, int c, int d, int cin) {
  return a ^ b ^ c ^ d ^ cin;
}

__device__ __forceinline__ void reduce4(const int Ab[4], const int Bb[4], int R[8]) {
  int G[4][8];
#pragma unroll
  for (int r = 0; r < 4; ++r)
#pragma unroll
    for (int c = 0; c < 8; ++c) G[r][c] = 0;
#pragma unroll
  for (int r = 0; r < 4; ++r)
#pragma unroll
    for (int k = 0; k < 4; ++k) G[r][(4 - r) + k] = Ab[k] & Bb[3 - r];

  R[7] = G[0][7] | G[1][7] | G[2][7] | G[3][7];
  R[6] = G[0][6] | G[1][6] | G[2][6] | G[3][6];
  int carry;
  comp1(G[0][5], G[1][5], G[2][5], G[3][5], R[5], carry);
  R[4] = comp2(G[0][4], G[1][4], G[2][4], G[3][4], carry);
  int c = 0;
#pragma unroll
  for (int col = 3; col >= 0; --col)
    badd4(G[0][col], G[1][col], G[2][col], G[3][col], c, c, R[col]);
}

__device__ __forceinline__ void two_row(int aH, int aL, int bH, int bL, int c0,
                                        int R[8]) {
#pragma unroll
  for (int i = 0; i < 8; ++i) R[i] = 0;
  int c = 0;
  badd2(aL & bL, 0,       c, c, R[c0 + 1]);
  badd2(aH & bL, aL & bH, c, c, R[c0]);
  badd2(0,       aH & bH, c, c, R[c0 - 1]);
  badd2(0,       0,       c, c, R[c0 - 2]);
}

__device__ int approx_mult8_dev(int a, int b) {
  int A[8], B[8];
#pragma unroll
  for (int i = 0; i < 8; ++i) {
    A[i] = (a >> (7 - i)) & 1;
    B[i] = (b >> (7 - i)) & 1;
  }
  int ALXL[8], AHXL[8], ALXH[8];
  reduce4(&A[4], &B[4], ALXL);
  reduce4(&A[0], &B[4], AHXL);
  reduce4(&A[4], &B[0], ALXH);
  int HH[8], HL[8], LH[8], LL[8];
  two_row(A[0], A[1], B[0], B[1], 2, HH);
  two_row(A[0], A[1], B[2], B[3], 4, HL);
  two_row(A[2], A[3], B[0], B[1], 4, LH);
  two_row(A[2], A[3], B[2], B[3], 6, LL);
  int R1[8];
  int c = 0;
#pragma unroll
  for (int col = 7; col >= 0; --col)
    badd4(HH[col], LH[col], HL[col], LL[col], c, c, R1[col]);
  int out = 0;
  c = 0;
#pragma unroll
  for (int i = 15; i >= 0; --i) {
    int s = c;
    if (i < 8) s += R1[i];
    if (i >= 8) s += ALXL[i - 8];
    if (i >= 4 && i < 12) s += ALXH[i - 4] + AHXL[i - 4];
    c = s >> 1;
    out += (s & 1) << (15 - i);
  }
  return out;
}

__global__ void __launch_bounds__(256)
approx_build_tables(const float* __restrict__ x, const float* __restrict__ w,
                    int* __restrict__ err, unsigned char* __restrict__ xq,
                    unsigned char* __restrict__ wq) {
  const int tid = blockIdx.x * 256 + threadIdx.x;
  const int a = tid >> 8, b = tid & 255;
  err[tid] = approx_mult8_dev(a, b) - a * b;

  int qw = (int)rintf(w[tid] * 255.0f);
  qw = qw < 0 ? 0 : (qw > 255 ? 255 : qw);
  wq[tid] = (unsigned char)qw;

  if (tid < 32 * 256) {
    int qx = (int)rintf(x[tid] * 255.0f);
    qx = qx < 0 ? 0 : (qx > 255 ? 255 : qx);
    xq[tid] = (unsigned char)qx;
  }
}

typedef __attribute__((ext_vector_type(8))) int v8i;

__global__ void __launch_bounds__(256)
approx_gemm_wmma(const unsigned char* __restrict__ xq,
                 const unsigned char* __restrict__ wq,
                 int* __restrict__ accI) {
  __shared__ __align__(16) unsigned char sX[32 * 256];
  __shared__ __align__(16) unsigned char sW[64 * 256];

  const int tid = threadIdx.x;
  const int nBlockBase = blockIdx.x * 64;

  {
    const uint32_t* gx = (const uint32_t*)xq;
    uint32_t* lx = (uint32_t*)sX;
    for (int i = tid; i < (32 * 256) / 4; i += 256) lx[i] = gx[i];
    const uint32_t* gw = (const uint32_t*)(wq + nBlockBase * 256);
    uint32_t* lw = (uint32_t*)sW;
    for (int i = tid; i < (64 * 256) / 4; i += 256) lw[i] = gw[i];
  }
  __syncthreads();

  const int wave = tid >> 5;
  const int lane = tid & 31;
  const int m0 = (wave & 1) * 16;
  const int n0l = (wave >> 1) * 16;

  const int laneLo  = lane & 15;
  const int laneHiA = (lane & 16) ? 8 : 0;
  const int laneHiB = (lane & 16) ? 16 : 0;

  const unsigned char* aRowFrag = sX + (m0 + laneLo) * 256;
  const unsigned char* bRowFrag = sW + (n0l + laneLo) * 256;

  v8i acc = {};
#pragma unroll
  for (int kc = 0; kc < 4; ++kc) {
    const int kb = kc * 64;
    v8i afrag, bfrag;
#pragma unroll
    for (int v = 0; v < 8; ++v) {
      const int ka = kb + ((v >> 1) << 4) + ((v & 1) << 2) + laneHiA;
      afrag[v] = (int)(*(const uint32_t*)(aRowFrag + ka));
      const int kbq = kb + ((v >> 1) << 4) + ((v & 1) << 2) + laneHiA;
      bfrag[v] = (int)(*(const uint32_t*)(bRowFrag + kbq));
    }
    acc = __builtin_amdgcn_wmma_i32_16x16x64_iu8(false, afrag, false, bfrag,
                                                 acc, false, false);
  }

  const int n = nBlockBase + n0l + laneLo;
#pragma unroll
  for (int j = 0; j < 8; ++j) {
    const int m = m0 + j + ((lane & 16) ? 8 : 0);
    accI[m * 256 + n] = acc[j];
  }
}

__global__ void __launch_bounds__(256)
approx_err_finalize(const unsigned char* __restrict__ xq,
                    const unsigned char* __restrict__ wq,
                    const int* __restrict__ err,
                    const int* __restrict__ accI, float* __restrict__ out) {
  __shared__ __align__(16) unsigned char sXrow[256];

  const int m = blockIdx.x;
  const int n = threadIdx.x;

  if (n < 64)
    ((uint32_t*)sXrow)[n] = ((const uint32_t*)(xq + m * 256))[n];
  __syncthreads();

  const unsigned char* wRow = wq + n * 256;
  int e = 0;
  for (int k = 0; k < 256; k += 4) {
    const uint32_t av = *(const uint32_t*)(sXrow + k);
    const uint32_t bv = *(const uint32_t*)(wRow + k);
    e += err[(( av         & 255u) << 8) | ( bv         & 255u)];
    e += err[(((av >> 8 )  & 255u) << 8) | ((bv >> 8 )  & 255u)];
    e += err[(((av >> 16)  & 255u) << 8) | ((bv >> 16)  & 255u)];
    e += err[(( av >> 24)          << 8) | ( bv >> 24)];
  }

  const int idx = m * 256 + n;
  out[idx] = (float)(accI[idx] + e) * (1.0f / (255.0f * 255.0f));
}

extern "C" void kernel_launch(void* const* d_in, const int* in_sizes, int n_in,
                              void* d_out, int out_size, void* d_ws, size_t ws_size,
                              hipStream_t stream) {
  const float* x = (const float*)d_in[0];
  const float* w = (const float*)d_in[1];
  float* out = (float*)d_out;

  char* ws = (char*)d_ws;
  int* err = (int*)ws;
  unsigned char* xq = (unsigned char*)(ws + 262144);
  unsigned char* wq = (unsigned char*)(ws + 262144 + 8192);
  int* accI = (int*)(ws + 262144 + 8192 + 65536);

  approx_build_tables<<<256, 256, 0, stream>>>(x, w, err, xq, wq);
  approx_gemm_wmma<<<4, 256, 0, stream>>>(xq, wq, accI);
  approx_err_finalize<<<32, 256, 0, stream>>>(xq, wq, err, accI, out);
}
